// ImageMoE_25537875542065
// MI455X (gfx1250) — compile-verified, matrix operand order checked by reading, not run on hardware
//
#include <hip/hip_runtime.h>
#include <hip/hip_bf16.h>
#include <stdint.h>

typedef _Float16 half_t;
typedef __attribute__((ext_vector_type(16))) _Float16 v16h;
typedef __attribute__((ext_vector_type(8)))  _Float16 v8h;
typedef __attribute__((ext_vector_type(8)))  float    v8f;

#define TOKENS   12544      // 64 * 196
#define DMODEL   768
#define NHEADS   8
#define HEADDIM  96
#define SEQL     64         // attention runs over the batch axis (faithful quirk)
#define NPATCH   196
#define NEXPERT  10
#define EHID     256
#define PATCHD   256
#define IMGW     224

// ---------------------------------------------------------------------------
// CDNA5 async global->LDS copy (bypasses VGPRs, tracked by ASYNCcnt).
// LDS destination operand is the wave-relative LDS byte address = low 32 bits
// of the generic __shared__ pointer (aperture truncation rule).
// ---------------------------------------------------------------------------
__device__ inline void async_g2l_b128(const void* g, void* l) {
  asm volatile("global_load_async_to_lds_b128 %0, %1, off"
               :
               : "v"((uint32_t)(uintptr_t)l), "v"(g)
               : "memory");
}
__device__ inline void async_wait0() {
  asm volatile("s_wait_asynccnt 0" ::: "memory");
}

// ---------------------------------------------------------------------------
// WMMA fragment load: rows stored contiguous-K in LDS.
// lane 0-15  : halves 0-7 = K[kb+0..7],  halves 8-15 = K[kb+16..23]
// lane 16-31 : halves 0-7 = K[kb+8..15], halves 8-15 = K[kb+24..31]
// ---------------------------------------------------------------------------
__device__ inline v16h frag_ld(const half_t* rowp, int kb, int hi) {
  v8h lo = *(const v8h*)(rowp + kb + hi * 8);
  v8h hh = *(const v8h*)(rowp + kb + 16 + hi * 8);
  return __builtin_shufflevector(lo, hh, 0,1,2,3,4,5,6,7,8,9,10,11,12,13,14,15);
}

__device__ inline v8f wmma16(v16h a, v16h b, v8f c) {
  return __builtin_amdgcn_wmma_f32_16x16x32_f16(false, a, false, b, (short)0, c,
                                                false, false);
}

__device__ inline float wave_sum(float v) {
  #pragma unroll
  for (int off = 16; off > 0; off >>= 1) v += __shfl_xor(v, off, 32);
  return v;
}

// ---------------------------------------------------------------------------
// Generic f16 WMMA GEMM:  C(M,N) = A(M,K) * Bt(N,K)^T  (Bt pre-transposed)
// Epilogue: +bias[n], ReLU, optional pos_emb[(m%196)*N+n], per-row scale,
// f32 store or f32 accumulate, optional f16 store.
// ---------------------------------------------------------------------------
#define BM 128
#define BN 128
#define BK 32
#define LDT 40   // LDS row stride in halves (80B, 16B-aligned groups)

__global__ __launch_bounds__(256) void gemm_f16_kernel(
    const half_t* __restrict__ A, const half_t* __restrict__ Bt,
    const float* __restrict__ bias, float* __restrict__ Cf,
    half_t* __restrict__ Ch, const float* __restrict__ posAdd,
    const float* __restrict__ rowScale,
    int M, int N, int K, int doRelu, int accum)
{
  __shared__ __align__(16) half_t As[BM * LDT];
  __shared__ __align__(16) half_t Bs[BN * LDT];

  const int tid   = threadIdx.x;
  const int tileM = blockIdx.y * BM;
  const int tileN = blockIdx.x * BN;

  const int wave = tid >> 5;
  const int lane = tid & 31;
  const int lh   = lane & 15;
  const int hi   = lane >> 4;
  const int wr   = wave >> 2;   // 0..1 : 64-row strip
  const int wc   = wave & 3;    // 0..3 : 32-col strip

  v8f acc[4][2] = {};

  for (int kb = 0; kb < K; kb += BK) {
    // async-stage A/B tiles into LDS: 512 chunks of 8 halves per tile
    #pragma unroll
    for (int it = 0; it < 2; it++) {
      int c   = tid + it * 256;
      int row = c >> 2;
      int kg  = (c & 3) << 3;
      int ga  = tileM + row; if (ga > M - 1) ga = M - 1;  // clamp (masked on store)
      async_g2l_b128(A  + (size_t)ga  * K + kb + kg, As + row * LDT + kg);
      int gbr = tileN + row;                               // N % 128 == 0 always
      async_g2l_b128(Bt + (size_t)gbr * K + kb + kg, Bs + row * LDT + kg);
    }
    if (kb + BK < K) {  // hint next K tile -> global_prefetch_b8
      int pr = tileM + (tid >> 2); if (pr > M - 1) pr = M - 1;
      __builtin_prefetch(A  + (size_t)pr * K + kb + BK, 0, 1);
      __builtin_prefetch(Bt + (size_t)(tileN + (tid >> 2)) * K + kb + BK, 0, 1);
    }
    async_wait0();
    __syncthreads();

    v16h af[4], bf[2];
    #pragma unroll
    for (int i = 0; i < 4; i++)
      af[i] = frag_ld(As + (wr * 64 + i * 16 + lh) * LDT, 0, hi);
    #pragma unroll
    for (int j = 0; j < 2; j++)
      bf[j] = frag_ld(Bs + (wc * 32 + j * 16 + lh) * LDT, 0, hi);

    #pragma unroll
    for (int i = 0; i < 4; i++)
      #pragma unroll
      for (int j = 0; j < 2; j++)
        acc[i][j] = wmma16(af[i], bf[j], acc[i][j]);
    __syncthreads();
  }

  // epilogue (C layout: lanes 0-15 rows 0-7, lanes 16-31 rows 8-15; col = lane&15)
  #pragma unroll
  for (int i = 0; i < 4; i++) {
    #pragma unroll
    for (int j = 0; j < 2; j++) {
      int col  = tileN + wc * 32 + j * 16 + lh;
      int rbase = tileM + wr * 64 + i * 16 + hi * 8;
      float bcol = bias ? bias[col] : 0.f;
      #pragma unroll
      for (int r = 0; r < 8; r++) {
        int rr = rbase + r;
        if (rr < M) {
          float v = acc[i][j][r] + bcol;
          if (doRelu) v = fmaxf(v, 0.f);
          if (posAdd) v += posAdd[(size_t)(rr % NPATCH) * N + col];
          if (rowScale) v *= rowScale[rr];
          size_t o = (size_t)rr * N + col;
          if (Cf) { if (accum) Cf[o] += v; else Cf[o] = v; }
          if (Ch) Ch[o] = (half_t)v;
        }
      }
    }
  }
}

// ---------------------------------------------------------------------------
// Fused attention for one (patch n, head h): S = q k^T/sqrt(96), softmax,
// O = P v. Writes full attn probs (f32) and context (f16).
// ---------------------------------------------------------------------------
__global__ __launch_bounds__(256) void attn_kernel(
    const half_t* __restrict__ Q, const half_t* __restrict__ Kx,
    const half_t* __restrict__ V, float* __restrict__ attnBuf,
    half_t* __restrict__ attnOut)
{
  __shared__ __align__(16) char smem[26624 + 13824 + 9216];
  half_t* qs = (half_t*)smem;                      // 64 x 104
  half_t* ks = qs + 64 * 104;                      // 64 x 104
  float*  ss = (float*)smem;                       // 64 x 64 (overlays q/k later)
  half_t* vt = (half_t*)(smem + 26624);            // 96 x 72  (d-major = N x K)
  half_t* ps = (half_t*)(smem + 26624 + 13824);    // 64 x 72

  const int n = blockIdx.x, h = blockIdx.y, tid = threadIdx.x;

  for (int c = tid; c < 64 * 12; c += 256) {       // q,k rows: async 8-half chunks
    int l = c / 12, dg = (c % 12) << 3;
    size_t base = (size_t)(l * NPATCH + n) * DMODEL + h * HEADDIM + dg;
    async_g2l_b128(Q  + base, qs + l * 104 + dg);
    async_g2l_b128(Kx + base, ks + l * 104 + dg);
  }
  for (int c = tid; c < 64 * 96; c += 256) {       // v transposed (scalar path)
    int l = c / 96, d = c % 96;
    vt[d * 72 + l] = V[(size_t)(l * NPATCH + n) * DMODEL + h * HEADDIM + d];
  }
  async_wait0();
  __syncthreads();

  const int wave = tid >> 5, lane = tid & 31, lh = lane & 15, hi = lane >> 4;

  { // S = q k^T  (64x64, K=96)
    int c4 = wave & 3;
    int r0 = (wave >> 2) * 2;
    v8f a0 = {}, a1 = {};
    #pragma unroll
    for (int kb = 0; kb < 96; kb += 32) {
      v16h bq  = frag_ld(ks + (c4 * 16 + lh) * 104, kb, hi);
      v16h aq0 = frag_ld(qs + (r0 * 16 + lh) * 104, kb, hi);
      v16h aq1 = frag_ld(qs + ((r0 + 1) * 16 + lh) * 104, kb, hi);
      a0 = wmma16(aq0, bq, a0);
      a1 = wmma16(aq1, bq, a1);
    }
    __syncthreads();                                // q/k dead -> ss may overlay
    const float sc = 0.1020620726159658f;           // 1/sqrt(96)
    #pragma unroll
    for (int r = 0; r < 8; r++) {
      int col = c4 * 16 + lh;
      ss[(r0 * 16 + hi * 8 + r) * 64 + col]       = a0[r] * sc;
      ss[((r0 + 1) * 16 + hi * 8 + r) * 64 + col] = a1[r] * sc;
    }
  }
  __syncthreads();

  if (tid < 64) {                                   // softmax row tid
    int l = tid;
    float mx = -3.4e38f;
    for (int s = 0; s < 64; s++) mx = fmaxf(mx, ss[l * 64 + s]);
    float sum = 0.f;
    for (int s = 0; s < 64; s++) {
      float e = __expf(ss[l * 64 + s] - mx);
      ss[l * 64 + s] = e; sum += e;
    }
    float inv = 1.f / sum;
    float* ab = attnBuf + (size_t)(n * 8 + h) * 4096 + l * 64;
    for (int s = 0; s < 64; s++) {
      float pr = ss[l * 64 + s] * inv;
      ab[s] = pr;
      ps[l * 72 + s] = (half_t)pr;
    }
  }
  __syncthreads();

  { // O = P v  (64x96, K=64)
    int r = wave >> 1, g = wave & 1;
    v8f o[3] = {};
    #pragma unroll
    for (int kb = 0; kb < 64; kb += 32) {
      v16h ap = frag_ld(ps + (r * 16 + lh) * 72, kb, hi);
      #pragma unroll
      for (int j = 0; j < 3; j++) {
        v16h bv = frag_ld(vt + ((g * 3 + j) * 16 + lh) * 72, kb, hi);
        o[j] = wmma16(ap, bv, o[j]);
      }
    }
    #pragma unroll
    for (int j = 0; j < 3; j++) {
      int d = (g * 3 + j) * 16 + lh;
      #pragma unroll
      for (int rr = 0; rr < 8; rr++) {
        int l = r * 16 + hi * 8 + rr;
        attnOut[(size_t)(l * NPATCH + n) * DMODEL + h * HEADDIM + d] = (half_t)o[j][rr];
      }
    }
  }
}

// ---------------------------------------------------------------------------
// small helper kernels
// ---------------------------------------------------------------------------
__global__ void transpose_f16_kernel(const float* __restrict__ in,
                                     half_t* __restrict__ out, int K, int N) {
  int e = blockIdx.y;
  size_t off = (size_t)e * K * N;
  int i = blockIdx.x * 256 + threadIdx.x;
  if (i >= K * N) return;
  int nn = i / K, kk = i % K;
  out[off + i] = (half_t)in[off + (size_t)kk * N + nn];  // out[n*K+k] = in[k*N+n]
}

__global__ void patchify_kernel(const float* __restrict__ x,
                                half_t* __restrict__ outh) {
  int i = blockIdx.x * 256 + threadIdx.x;
  if (i >= TOKENS * PATCHD) return;
  int t = i >> 8, p = i & 255;
  int bb = t / NPATCH, s = t % NPATCH;
  int ph = s / 14, pw = s % 14;
  int pr = p >> 4, pc = p & 15;
  outh[i] = (half_t)x[(size_t)bb * IMGW * IMGW + (ph * 16 + pr) * IMGW + pw * 16 + pc];
}

__global__ void zero_kernel(float* p, int n) {
  int i = blockIdx.x * 256 + threadIdx.x;
  if (i < n) p[i] = 0.f;
}

__global__ void headmean_kernel(const float* __restrict__ attnBuf,
                                float* __restrict__ meanAttn) {
  int i = blockIdx.x * 256 + threadIdx.x;
  if (i >= NPATCH * 4096) return;
  int nn = i >> 12, rem = i & 4095;
  float s = 0.f;
  #pragma unroll
  for (int h = 0; h < 8; h++) s += attnBuf[(size_t)(nn * 8 + h) * 4096 + rem];
  meanAttn[i] = s * 0.125f;
}

// aw[b,s] = mean over 64 of meanAttn.flat viewed as (64,196,64)  (.view-faithful)
__global__ void aw_kernel(const float* __restrict__ meanAttn,
                          float* __restrict__ aw) {
  int t = blockIdx.x * 256 + threadIdx.x;
  if (t >= TOKENS) return;
  int bb = t / NPATCH, s = t % NPATCH;
  const float* p = meanAttn + (size_t)bb * NPATCH * 64 + s * 64;
  float acc = 0.f;
  for (int k = 0; k < 64; k++) acc += p[k];
  aw[t] = acc * (1.f / 64.f);
}

// one wave per token: gate logits, softmax top-2 renormalized -> wT[e][t]
__global__ __launch_bounds__(256) void gate_kernel(
    const float* __restrict__ X, const float* __restrict__ gW,
    const float* __restrict__ gb, float* __restrict__ wT)
{
  int wave = threadIdx.x >> 5, lane = threadIdx.x & 31;
  int t = blockIdx.x * 8 + wave;
  if (t >= TOKENS) return;
  float acc[NEXPERT];
  #pragma unroll
  for (int e = 0; e < NEXPERT; e++) acc[e] = 0.f;
  for (int d = lane; d < DMODEL; d += 32) {
    float xv = X[(size_t)t * DMODEL + d];
    #pragma unroll
    for (int e = 0; e < NEXPERT; e++) acc[e] += xv * gW[d * NEXPERT + e];
  }
  #pragma unroll
  for (int e = 0; e < NEXPERT; e++) acc[e] = wave_sum(acc[e]);
  if (lane == 0) {
    float l0 = -3.4e38f; int i0 = 0;
    #pragma unroll
    for (int e = 0; e < NEXPERT; e++) {
      float v = acc[e] + gb[e]; acc[e] = v;
      if (v > l0) { l0 = v; i0 = e; }
    }
    float l1 = -3.4e38f; int i1 = 0;
    #pragma unroll
    for (int e = 0; e < NEXPERT; e++)
      if (e != i0 && acc[e] > l1) { l1 = acc[e]; i1 = e; }
    float w0 = 1.f / (1.f + __expf(l1 - l0));   // normalized top-2 pair
    wT[(size_t)i0 * TOKENS + t] = w0;
    wT[(size_t)i1 * TOKENS + t] = 1.f - w0;
  }
}

// one wave per token: LayerNorm * lng + lnb, then * aw[t]  -> f16
__global__ __launch_bounds__(256) void ln_kernel(
    const float* __restrict__ X, const float* __restrict__ g,
    const float* __restrict__ b, const float* __restrict__ aw,
    half_t* __restrict__ outh)
{
  int wave = threadIdx.x >> 5, lane = threadIdx.x & 31;
  int t = blockIdx.x * 8 + wave;
  if (t >= TOKENS) return;
  const float* row = X + (size_t)t * DMODEL;
  float s1 = 0.f, s2 = 0.f;
  for (int d = lane; d < DMODEL; d += 32) { float v = row[d]; s1 += v; s2 += v * v; }
  s1 = wave_sum(s1); s2 = wave_sum(s2);
  float mu = s1 * (1.f / DMODEL);
  float var = s2 * (1.f / DMODEL) - mu * mu;
  float rinv = rsqrtf(var + 1e-5f);
  float a = aw[t];
  half_t* orow = outh + (size_t)t * DMODEL;
  for (int d = lane; d < DMODEL; d += 32)
    orow[d] = (half_t)(((row[d] - mu) * rinv * g[d] + b[d]) * a);
}

__global__ void gv_kernel(const float* __restrict__ sv, const float* __restrict__ aw,
                          float* __restrict__ gvf, half_t* __restrict__ gvh) {
  int d = blockIdx.x * 256 + threadIdx.x;
  int bb = blockIdx.y;
  if (d >= DMODEL) return;
  float acc = 0.f;
  for (int s = 0; s < NPATCH; s++) {
    int t = bb * NPATCH + s;
    acc += sv[(size_t)t * DMODEL + d] * aw[t];
  }
  gvf[bb * DMODEL + d] = acc;
  gvh[bb * DMODEL + d] = (half_t)acc;
}

// ---------------------------------------------------------------------------
extern "C" void kernel_launch(void* const* d_in, const int* in_sizes, int n_in,
                              void* d_out, int out_size, void* d_ws, size_t ws_size,
                              hipStream_t stream) {
  (void)in_sizes; (void)n_in; (void)out_size; (void)ws_size;
  const float* x       = (const float*)d_in[0];
  const float* pos_emb = (const float*)d_in[1];
  const float* patch_W = (const float*)d_in[2];
  const float* patch_b = (const float*)d_in[3];
  const float* vec_W   = (const float*)d_in[40];
  const float* vec_b   = (const float*)d_in[41];
  const float* cls_W   = (const float*)d_in[42];
  const float* cls_b   = (const float*)d_in[43];

  const size_t TD = (size_t)TOKENS * DMODEL;

  char* p = (char*)d_ws;
  auto carve = [&](size_t bytes) -> char* {
    char* r = p; p += (bytes + 255) & ~(size_t)255; return r;
  };
  // f16 weight staging (transposed N x K)
  half_t* patchWt = (half_t*)carve((size_t)768 * 256 * 2);
  half_t* pWt     = (half_t*)carve((size_t)768 * 768 * 2);
  half_t* WqT     = (half_t*)carve((size_t)768 * 768 * 2);
  half_t* WkT     = (half_t*)carve((size_t)768 * 768 * 2);
  half_t* WvT     = (half_t*)carve((size_t)768 * 768 * 2);
  half_t* WoT     = (half_t*)carve((size_t)768 * 768 * 2);
  half_t* eW1t    = (half_t*)carve((size_t)NEXPERT * 256 * 768 * 2);
  half_t* eW2t    = (half_t*)carve((size_t)NEXPERT * 768 * 256 * 2);
  half_t* vecWt   = (half_t*)carve((size_t)768 * 768 * 2);
  half_t* clsWt   = (half_t*)carve((size_t)768 * 768 * 2);
  // f16 activations
  half_t* patches = (half_t*)carve((size_t)TOKENS * PATCHD * 2);
  half_t* inH     = (half_t*)carve(TD * 2);
  half_t* projH   = (half_t*)carve(TD * 2);
  half_t* QH      = (half_t*)carve(TD * 2);
  half_t* KH      = (half_t*)carve(TD * 2);
  half_t* VH      = (half_t*)carve(TD * 2);
  half_t* aoH     = (half_t*)carve(TD * 2);
  half_t* YH      = (half_t*)carve(TD * 2);
  half_t* HH      = (half_t*)carve((size_t)TOKENS * EHID * 2);
  half_t* lnH     = (half_t*)carve(TD * 2);
  half_t* gvH     = (half_t*)carve((size_t)64 * DMODEL * 2);
  // f32 buffers
  float* Yf       = (float*)carve(TD * 4);
  float* attnBuf  = (float*)carve((size_t)NPATCH * 8 * 4096 * 4);
  float* meanAttn = (float*)carve((size_t)NPATCH * 4096 * 4);
  float* awBuf    = (float*)carve((size_t)TOKENS * 4);
  float* wT       = (float*)carve((size_t)NEXPERT * TOKENS * 4);
  float* OutAcc   = (float*)carve(TD * 4);

  float* out       = (float*)d_out;
  float* firstVec  = out;
  float* secondVec = out + TD;
  float* gvOut     = secondVec + TD;
  float* logits    = gvOut + (size_t)64 * DMODEL;

  auto convT = [&](const float* src, half_t* dst, int K, int N, int count) {
    dim3 g((K * N + 255) / 256, count);
    transpose_f16_kernel<<<g, 256, 0, stream>>>(src, dst, K, N);
  };
  auto gemm = [&](const half_t* A, const half_t* Bt, const float* bias,
                  float* Cf, half_t* Ch, const float* posAdd, const float* rs,
                  int M, int N, int K, int relu, int accum) {
    dim3 g(N / BN, (M + BM - 1) / BM);
    gemm_f16_kernel<<<g, 256, 0, stream>>>(A, Bt, bias, Cf, Ch, posAdd, rs,
                                           M, N, K, relu, accum);
  };

  convT(patch_W, patchWt, 256, 768, 1);
  convT(vec_W,   vecWt,   768, 768, 1);
  convT(cls_W,   clsWt,   768, 768, 1);

  patchify_kernel<<<(TOKENS * PATCHD + 255) / 256, 256, 0, stream>>>(x, patches);
  // patch embed + bias + pos_emb
  gemm(patches, patchWt, patch_b, nullptr, inH, pos_emb, nullptr,
       TOKENS, DMODEL, PATCHD, 0, 0);

  for (int layer = 0; layer < 2; layer++) {
    void* const* mp = d_in + (layer == 0 ? 4 : 22);
    const float* pW  = (const float*)mp[0];  const float* pb  = (const float*)mp[1];
    const float* Wq  = (const float*)mp[2];  const float* bq  = (const float*)mp[3];
    const float* Wk  = (const float*)mp[4];  const float* bk  = (const float*)mp[5];
    const float* Wv  = (const float*)mp[6];  const float* bv  = (const float*)mp[7];
    const float* Wo  = (const float*)mp[8];  const float* bo  = (const float*)mp[9];
    const float* gW  = (const float*)mp[10]; const float* gb  = (const float*)mp[11];
    const float* eW1 = (const float*)mp[12]; const float* eb1 = (const float*)mp[13];
    const float* eW2 = (const float*)mp[14]; const float* eb2 = (const float*)mp[15];
    const float* lng = (const float*)mp[16]; const float* lnb = (const float*)mp[17];

    convT(pW, pWt, 768, 768, 1);
    convT(Wq, WqT, 768, 768, 1);
    convT(Wk, WkT, 768, 768, 1);
    convT(Wv, WvT, 768, 768, 1);
    convT(Wo, WoT, 768, 768, 1);
    convT(eW1, eW1t, 768, 256, NEXPERT);   // per-expert (256 x 768)
    convT(eW2, eW2t, 256, 768, NEXPERT);   // per-expert (768 x 256)

    // input projection, then q/k/v
    gemm(inH,   pWt, pb, nullptr, projH, nullptr, nullptr, TOKENS, DMODEL, DMODEL, 0, 0);
    gemm(projH, WqT, bq, nullptr, QH,    nullptr, nullptr, TOKENS, DMODEL, DMODEL, 0, 0);
    gemm(projH, WkT, bk, nullptr, KH,    nullptr, nullptr, TOKENS, DMODEL, DMODEL, 0, 0);
    gemm(projH, WvT, bv, nullptr, VH,    nullptr, nullptr, TOKENS, DMODEL, DMODEL, 0, 0);

    attn_kernel<<<dim3(NPATCH, NHEADS), 256, 0, stream>>>(QH, KH, VH, attnBuf, aoH);

    // output projection -> Y (f32 for gate, f16 for experts)
    gemm(aoH, WoT, bo, Yf, YH, nullptr, nullptr, TOKENS, DMODEL, DMODEL, 0, 0);

    headmean_kernel<<<(NPATCH * 4096 + 255) / 256, 256, 0, stream>>>(attnBuf, meanAttn);
    aw_kernel<<<(TOKENS + 255) / 256, 256, 0, stream>>>(meanAttn, awBuf);

    zero_kernel<<<(NEXPERT * TOKENS + 255) / 256, 256, 0, stream>>>(wT, NEXPERT * TOKENS);
    gate_kernel<<<(TOKENS + 7) / 8, 256, 0, stream>>>(Yf, gW, gb, wT);

    zero_kernel<<<((int)TD + 255) / 256, 256, 0, stream>>>(OutAcc, (int)TD);
    for (int e = 0; e < NEXPERT; e++) {
      gemm(YH, eW1t + (size_t)e * 768 * 256, eb1 + e * EHID, nullptr, HH,
           nullptr, nullptr, TOKENS, EHID, DMODEL, 1, 0);                 // ReLU
      gemm(HH, eW2t + (size_t)e * 768 * 256, eb2 + e * DMODEL, OutAcc, nullptr,
           nullptr, wT + (size_t)e * TOKENS, TOKENS, DMODEL, EHID, 0, 1); // += w_e * (.)
    }

    ln_kernel<<<(TOKENS + 7) / 8, 256, 0, stream>>>(OutAcc, lng, lnb, awBuf, lnH);

    if (layer == 0) {
      gemm(lnH, vecWt, vec_b, firstVec, inH, nullptr, nullptr,
           TOKENS, DMODEL, DMODEL, 0, 0);   // f16 copy feeds moe2
    } else {
      gemm(lnH, vecWt, vec_b, secondVec, nullptr, nullptr, nullptr,
           TOKENS, DMODEL, DMODEL, 0, 0);
    }
  }

  gv_kernel<<<dim3((DMODEL + 255) / 256, 64), 256, 0, stream>>>(secondVec, awBuf,
                                                                gvOut, gvH);
  gemm(gvH, clsWt, cls_b, logits, nullptr, nullptr, nullptr, 64, DMODEL, DMODEL, 0, 0);
}
